// AttenLayer_42820823941242
// MI455X (gfx1250) — matrix operand order repaired and checked by reading, not compiled, not run on hardware
//
#include <hip/hip_runtime.h>
#include <hip/hip_fp16.h>

#define B_ 4
#define S_ 512
#define N_ 4096
#define D_ 512
#define NEGV (-9.0e15f)
#define LDK 40  // 32-k tile + 8 halfs pad (80B row stride: 64B data + 16B pad)

typedef __attribute__((ext_vector_type(16))) _Float16 v16h;
typedef __attribute__((ext_vector_type(8)))  _Float16 v8h;
typedef __attribute__((ext_vector_type(4)))  _Float16 v4h;
typedef __attribute__((ext_vector_type(8)))  float    v8f;
typedef __attribute__((ext_vector_type(4)))  unsigned int u32x4;
typedef __attribute__((ext_vector_type(8)))  int i32x8;
typedef __attribute__((ext_vector_type(4)))  int i32x4;

union Frag { v16h v; v8h h[2]; };

// ---- TDM: DMA a [tileRows x 32] f16 tile (row stride rowStrideElems) into LDS,
// ---- writing rows of 64B + 16B pad (matches LDK=40 halfs). Tracked by TENSORcnt.
__device__ __forceinline__ void tdm_load_tile_f16(const _Float16* gptr,
                                                  unsigned lds_off,
                                                  unsigned tileRows,
                                                  unsigned rowStrideElems) {
  const unsigned long long ga = (unsigned long long)(uintptr_t)gptr;
  u32x4 g0;
  g0[0] = 1u;                                        // count=1 valid user D#
  g0[1] = lds_off;                                   // LDS byte address
  g0[2] = (unsigned)(ga & 0xffffffffu);              // global_addr[31:0]
  g0[3] = (unsigned)((ga >> 32) & 0x01ffffffu) | (2u << 30);  // addr[56:32] | type=2
  const unsigned td0 = 1u << 30, td1 = 1u << 30;     // huge dims -> no OOB clipping
  i32x8 g1;
  g1[0] = (int)((1u << 16)      // data_size = 2 bytes
              | (1u << 20)      // pad_enable
              | (3u << 22)      // pad_interval: 64B of data ...
              | (3u << 25));    // pad_amount:   ... then 4 DWORDs (16B) pad
  g1[1] = (int)((td0 & 0xffffu) << 16);                       // barrier_addr=0 | td0.lo
  g1[2] = (int)(((td0 >> 16) & 0xffffu) | ((td1 & 0xffffu) << 16));
  g1[3] = (int)(((td1 >> 16) & 0xffffu) | (32u << 16));       // tile_dim0 = 32 elems
  g1[4] = (int)(tileRows & 0xffffu);                          // tile_dim1; tile_dim2=0
  g1[5] = (int)rowStrideElems;                                // tensor_dim0_stride lo32
  g1[6] = 0;
  g1[7] = 0;
  const i32x4 z4 = {0, 0, 0, 0};
  const i32x8 z8 = {0, 0, 0, 0, 0, 0, 0, 0};
  __builtin_amdgcn_tensor_load_to_lds(g0, g1, z4, z4, z8, 0);
}

// One K=32 step: wave computes a 64(m) x 32(n) region = 4x2 WMMA tiles.
// As/Bs: [row][k] tiles, k contiguous, row stride LDK halfs.
__device__ __forceinline__ void wmma_step(const _Float16* __restrict__ As,
                                          const _Float16* __restrict__ Bs,
                                          int mBase, int nBase, int lane,
                                          v8f acc[4][2]) {
  const int hl  = lane >> 4;
  const int l15 = lane & 15;
  Frag a[4], b[2];
#pragma unroll
  for (int i = 0; i < 4; ++i) {
    const _Float16* ap = As + (mBase + i * 16 + l15) * LDK + hl * 8;
    a[i].h[0] = *(const v8h*)(ap);
    a[i].h[1] = *(const v8h*)(ap + 16);
  }
#pragma unroll
  for (int j = 0; j < 2; ++j) {
    const _Float16* bp = Bs + (nBase + j * 16 + l15) * LDK + hl * 8;
    b[j].h[0] = *(const v8h*)(bp);
    b[j].h[1] = *(const v8h*)(bp + 16);
  }
#pragma unroll
  for (int i = 0; i < 4; ++i)
#pragma unroll
    for (int j = 0; j < 2; ++j)
      acc[i][j] = __builtin_amdgcn_wmma_f32_16x16x32_f16(
          false, a[i].v, false, b[j].v, (short)0, acc[i][j], false, false);
}

// ---------------- Stage 1: Q/K/V = x^T W + b (f16 to ws; V stored transposed) ----------------
__global__ __launch_bounds__(256) void qkv_kernel(
    const float* __restrict__ x,
    const float* __restrict__ Wq, const float* __restrict__ bq,
    const float* __restrict__ Wk, const float* __restrict__ bk,
    const float* __restrict__ Wv, const float* __restrict__ bv,
    _Float16* __restrict__ Qh, _Float16* __restrict__ Kh, _Float16* __restrict__ Vt) {
  __shared__ __align__(16) _Float16 As[128 * LDK];
  __shared__ __align__(16) _Float16 Bs[128 * LDK];
  const int tid = threadIdx.x, lane = tid & 31, wid = tid >> 5;
  const int d0   = blockIdx.x * 128;
  const int row0 = blockIdx.y * 128;   // flat (b*N + n)
  const int b    = row0 / N_;
  const int n0   = row0 % N_;
  const float* W; const float* bias; _Float16* Out;
  if (blockIdx.z == 0)      { W = Wq; bias = bq; Out = Qh; }
  else if (blockIdx.z == 1) { W = Wk; bias = bk; Out = Kh; }
  else                      { W = Wv; bias = bv; Out = Vt; }
  const bool transposed = (blockIdx.z == 2);

  v8f acc[4][2];
  const v8f zero = {};
#pragma unroll
  for (int i = 0; i < 4; ++i)
#pragma unroll
    for (int j = 0; j < 2; ++j) acc[i][j] = zero;

  const float* xb = x + (size_t)b * S_ * N_;
  for (int ks = 0; ks < S_; ks += 32) {
    if (ks + 32 < S_) {
      __builtin_prefetch(&xb[(size_t)(ks + 32) * N_ + n0], 0, 3);
      __builtin_prefetch(&W[(size_t)(ks + 32) * D_ + d0], 0, 3);
    }
    // A tile: As[m][k] = x[b][ks+k][n0+m] (float4 along m)
    for (int idx = tid; idx < 1024; idx += 256) {
      int mq = idx & 31, k = idx >> 5, m = mq * 4;
      const float4 f = *(const float4*)&xb[(size_t)(ks + k) * N_ + n0 + m];
      As[(m + 0) * LDK + k] = (_Float16)f.x;
      As[(m + 1) * LDK + k] = (_Float16)f.y;
      As[(m + 2) * LDK + k] = (_Float16)f.z;
      As[(m + 3) * LDK + k] = (_Float16)f.w;
    }
    // B tile: Bs[d][k] = W[ks+k][d0+d] (float4 along d)
    for (int idx = tid; idx < 1024; idx += 256) {
      int dq = idx & 31, k = idx >> 5, d = dq * 4;
      const float4 f = *(const float4*)&W[(size_t)(ks + k) * D_ + d0 + d];
      Bs[(d + 0) * LDK + k] = (_Float16)f.x;
      Bs[(d + 1) * LDK + k] = (_Float16)f.y;
      Bs[(d + 2) * LDK + k] = (_Float16)f.z;
      Bs[(d + 3) * LDK + k] = (_Float16)f.w;
    }
    __syncthreads();
    wmma_step(As, Bs, (wid & 1) * 64, (wid >> 1) * 32, lane, acc);
    __syncthreads();
  }
  const int hl = lane >> 4, l15 = lane & 15;
#pragma unroll
  for (int i = 0; i < 4; ++i)
#pragma unroll
    for (int j = 0; j < 2; ++j) {
      const int col = d0 + (wid >> 1) * 32 + j * 16 + l15;
      const float bb = bias[col];
#pragma unroll
      for (int e = 0; e < 8; ++e) {
        const int r = row0 + (wid & 1) * 64 + i * 16 + e + hl * 8;  // b*N + n
        const _Float16 h = (_Float16)(acc[i][j][e] + bb);
        if (!transposed) Out[(size_t)r * D_ + col] = h;            // Q,K: (B*N, D)
        else             Out[(size_t)col * (B_ * N_) + r] = h;     // Vt:  (D, B*N)
      }
    }
}

// -------- Stage 2: E = Q K^T, adjacency-masked. Q/K tiles via TDM, double-buffered. --------
__global__ __launch_bounds__(256) void scores_kernel(
    const _Float16* __restrict__ Qh, const _Float16* __restrict__ Kh,
    const int* __restrict__ adj, float* __restrict__ outA) {
  __shared__ __align__(16) _Float16 As[2][128 * LDK];
  __shared__ __align__(16) _Float16 Bs[2][128 * LDK];
  const int tid = threadIdx.x, lane = tid & 31, wid = tid >> 5;
  const int m0 = blockIdx.x * 128;  // key cols
  const int n0 = blockIdx.y * 128;  // query rows
  const int b  = blockIdx.z;
  const _Float16* Qt = Qh + (size_t)b * N_ * D_ + (size_t)n0 * D_;
  const _Float16* Kt = Kh + (size_t)b * N_ * D_ + (size_t)m0 * D_;
  const unsigned asOff0 = (unsigned)(uintptr_t)(void*)&As[0][0];
  const unsigned asOff1 = (unsigned)(uintptr_t)(void*)&As[1][0];
  const unsigned bsOff0 = (unsigned)(uintptr_t)(void*)&Bs[0][0];
  const unsigned bsOff1 = (unsigned)(uintptr_t)(void*)&Bs[1][0];

  v8f acc[4][2];
  const v8f zero = {};
#pragma unroll
  for (int i = 0; i < 4; ++i)
#pragma unroll
    for (int j = 0; j < 2; ++j) acc[i][j] = zero;

  if (wid == 0) {
    tdm_load_tile_f16(Qt, asOff0, 128, D_);
    tdm_load_tile_f16(Kt, bsOff0, 128, D_);
  }
  int p = 0;
  for (int ks = 0; ks < D_; ks += 32) {
    if (wid == 0) __builtin_amdgcn_s_wait_tensorcnt(0);
    __syncthreads();  // buf[p] ready; everyone done reading buf[p^1]
    if (wid == 0 && ks + 32 < D_) {
      tdm_load_tile_f16(Qt + ks + 32, p ? asOff0 : asOff1, 128, D_);
      tdm_load_tile_f16(Kt + ks + 32, p ? bsOff0 : bsOff1, 128, D_);
    }
    wmma_step(As[p], Bs[p], (wid & 1) * 64, (wid >> 1) * 32, lane, acc);
    p ^= 1;
  }
  float* Eb = outA + (size_t)b * N_ * N_;
  const int hl = lane >> 4, l15 = lane & 15;
#pragma unroll
  for (int i = 0; i < 4; ++i)
#pragma unroll
    for (int j = 0; j < 2; ++j) {
      const int col = m0 + (wid >> 1) * 32 + j * 16 + l15;
#pragma unroll
      for (int e = 0; e < 8; ++e) {
        const int r = n0 + (wid & 1) * 64 + i * 16 + e + hl * 8;
        float v = acc[i][j][e];
        v = (adj[(size_t)r * N_ + col] > 0) ? v : NEGV;
        Eb[(size_t)r * N_ + col] = v;
      }
    }
}

// ------------- Stage 3: softmax over axis=1 (columns of E), in place -------------
__global__ __launch_bounds__(256) void softmax_kernel(float* __restrict__ att) {
  const int tid = threadIdx.x;
  const int c = tid & 63, ph = tid >> 6;
  const int col = blockIdx.x * 64 + c;
  float* E = att + (size_t)blockIdx.y * N_ * N_;
  float mx = -3.4e38f, sm = 0.0f;
  for (int n = ph; n < N_; n += 4) {
    float v  = E[(size_t)n * N_ + col];
    float nm = fmaxf(mx, v);
    sm = sm * __expf(mx - nm) + __expf(v - nm);
    mx = nm;
  }
  __shared__ float smx[4][64], ssm[4][64];
  smx[ph][c] = mx; ssm[ph][c] = sm;
  __syncthreads();
  if (ph == 0) {
    float M = smx[0][c], S = ssm[0][c];
#pragma unroll
    for (int p = 1; p < 4; ++p) {
      float m2 = smx[p][c], s2 = ssm[p][c];
      float nm = fmaxf(M, m2);
      S = S * __expf(M - nm) + s2 * __expf(m2 - nm);
      M = nm;
    }
    smx[0][c] = M; ssm[0][c] = 1.0f / S;
  }
  __syncthreads();
  const float M = smx[0][c], inv = ssm[0][c];
  for (int n = ph; n < N_; n += 4) {
    const size_t idx = (size_t)n * N_ + col;
    E[idx] = __expf(E[idx] - M) * inv;
  }
}

// ----- Stage 4: V' = atten @ V / sqrt(512). V tile via TDM from Vt, double-buffered. -----
__global__ __launch_bounds__(256) void out_kernel(
    const float* __restrict__ att, const _Float16* __restrict__ Vt,
    float* __restrict__ outV) {
  __shared__ __align__(16) _Float16 As[2][128 * LDK];
  __shared__ __align__(16) _Float16 Bs[2][128 * LDK];
  const int tid = threadIdx.x, lane = tid & 31, wid = tid >> 5;
  const int d0 = blockIdx.x * 128;
  const int n0 = blockIdx.y * 128;
  const int b  = blockIdx.z;
  const float*    Ab = att + (size_t)b * N_ * N_;
  // Vt is (D, B*N): row d, col (b*N + m) -> k contiguous, TDM-able
  const _Float16* Vb = Vt + (size_t)d0 * (B_ * N_) + (size_t)b * N_;
  const unsigned bsOff0 = (unsigned)(uintptr_t)(void*)&Bs[0][0];
  const unsigned bsOff1 = (unsigned)(uintptr_t)(void*)&Bs[1][0];

  v8f acc[4][2];
  const v8f zero = {};
#pragma unroll
  for (int i = 0; i < 4; ++i)
#pragma unroll
    for (int j = 0; j < 2; ++j) acc[i][j] = zero;

  if (wid == 0) tdm_load_tile_f16(Vb, bsOff0, 128, B_ * N_);
  int p = 0;
  for (int ks = 0; ks < N_; ks += 32) {
    if (ks + 32 < N_) __builtin_prefetch(&Ab[(size_t)n0 * N_ + ks + 32], 0, 3);
    // A: atten fp32 -> f16 (float4 loads, ds_store_b64 packed stores)
    for (int idx = tid; idx < 1024; idx += 256) {
      int kq = idx & 7, r = idx >> 3;
      const float4 f = *(const float4*)&Ab[(size_t)(n0 + r) * N_ + ks + kq * 4];
      v4h h;
      h[0] = (_Float16)f.x; h[1] = (_Float16)f.y;
      h[2] = (_Float16)f.z; h[3] = (_Float16)f.w;
      *(v4h*)&As[p][r * LDK + kq * 4] = h;
    }
    if (wid == 0) __builtin_amdgcn_s_wait_tensorcnt(0);
    __syncthreads();  // As[p] staged + Bs[p] DMA'd; all readers of buf[p^1] done
    if (wid == 0 && ks + 32 < N_)
      tdm_load_tile_f16(Vb + ks + 32, p ? bsOff0 : bsOff1, 128, B_ * N_);
    wmma_step(As[p], Bs[p], (wid & 1) * 64, (wid >> 1) * 32, lane, acc);
    p ^= 1;
  }
  const float scale = 0.044194173824159216f;  // 1/sqrt(512)
  const int hl = lane >> 4, l15 = lane & 15;
#pragma unroll
  for (int i = 0; i < 4; ++i)
#pragma unroll
    for (int j = 0; j < 2; ++j) {
      const int col = d0 + (wid >> 1) * 32 + j * 16 + l15;
#pragma unroll
      for (int e = 0; e < 8; ++e) {
        const int r = n0 + (wid & 1) * 64 + i * 16 + e + hl * 8;
        outV[(size_t)(b * N_ + r) * D_ + col] = acc[i][j][e] * scale;
      }
    }
}

extern "C" void kernel_launch(void* const* d_in, const int* in_sizes, int n_in,
                              void* d_out, int out_size, void* d_ws, size_t ws_size,
                              hipStream_t stream) {
  (void)in_sizes; (void)n_in; (void)out_size; (void)ws_size;
  const float* x   = (const float*)d_in[0];
  const int*   adj = (const int*)d_in[1];
  const float* Wq  = (const float*)d_in[2];
  const float* bq  = (const float*)d_in[3];
  const float* Wk  = (const float*)d_in[4];
  const float* bk  = (const float*)d_in[5];
  const float* Wv  = (const float*)d_in[6];
  const float* bv  = (const float*)d_in[7];

  float* outV = (float*)d_out;                 // (B, N, D)
  float* outA = outV + (size_t)B_ * N_ * D_;   // (B, N, N) atten

  _Float16* Qh = (_Float16*)d_ws;              // (B*N, D) f16
  _Float16* Kh = Qh + (size_t)B_ * N_ * D_;    // (B*N, D) f16
  _Float16* Vt = Kh + (size_t)B_ * N_ * D_;    // (D, B*N) f16 (transposed V)

  qkv_kernel<<<dim3(D_ / 128, (B_ * N_) / 128, 3), 256, 0, stream>>>(
      x, Wq, bq, Wk, bk, Wv, bv, Qh, Kh, Vt);
  scores_kernel<<<dim3(N_ / 128, N_ / 128, B_), 256, 0, stream>>>(Qh, Kh, adj, outA);
  softmax_kernel<<<dim3(N_ / 64, B_), 256, 0, stream>>>(outA);
  out_kernel<<<dim3(D_ / 128, N_ / 128, B_), 256, 0, stream>>>(outA, Vt, outV);
}
